// EffAttention_1571958031010
// MI455X (gfx1250) — matrix operand order repaired and checked by reading, not compiled, not run on hardware
//
#include <hip/hip_runtime.h>
#include <hip/hip_bf16.h>

// ---------------------------------------------------------------------------
// EffAttention for MI455X (gfx1250, wave32, WMMA bf16).
//   qkv   = x @ (reduce_w @ qkv_w)              (weights fused algebraically)
//   attn  = softmax(Q K^T * 0.125) V            (block-diagonal, 256-tok blocks)
//   out   = o @ proj_w + proj_b
// All GEMMs via v_wmma_f32_16x16x32_bf16 (fp32 accumulate).
// ---------------------------------------------------------------------------

typedef __attribute__((ext_vector_type(16))) __bf16 v16bf;
typedef __attribute__((ext_vector_type(8)))  float  v8f;

union Frag16 { v16bf v; unsigned int u[8]; };

__device__ __forceinline__ unsigned short f32_to_bf16(float f) {
  unsigned int u = __float_as_uint(f);
  u += 0x7FFFu + ((u >> 16) & 1u);   // round-to-nearest-even
  return (unsigned short)(u >> 16);
}

// ---------------------------------------------------------------------------
// Kernel 1: fuse reduce_w[512,256] @ qkv_w[256,768] -> bf16 Wf[512,768],
//           and convert proj_w[256,512] -> bf16.
// ---------------------------------------------------------------------------
__global__ void prep_weights(const float* __restrict__ reduce_w,
                             const float* __restrict__ qkv_w,
                             const float* __restrict__ proj_w,
                             unsigned short* __restrict__ Wf,
                             unsigned short* __restrict__ PW) {
  int id = blockIdx.x * blockDim.x + threadIdx.x;
  if (id < 512 * 768) {
    int row = id / 768, col = id % 768;
    float s = 0.f;
    for (int k = 0; k < 256; ++k)
      s = fmaf(reduce_w[row * 256 + k], qkv_w[k * 768 + col], s);
    Wf[id] = f32_to_bf16(s);
  } else {
    int j = id - 512 * 768;
    if (j < 256 * 512) PW[j] = f32_to_bf16(proj_w[j]);
  }
}

// ---------------------------------------------------------------------------
// Kernel 2: qkv[16384,768] (bf16) = x[16384,512] (f32) @ Wf[512,768] (bf16)
// 64x64 tile per 128-thread workgroup; each wave: 32x32 = 2x2 WMMA tiles.
// ---------------------------------------------------------------------------
__global__ __launch_bounds__(128) void gemm_qkv(
    const float* __restrict__ X, const unsigned short* __restrict__ W,
    unsigned short* __restrict__ Y) {
  __shared__ unsigned short As[64][36];  // [m][k], bf16, padded (72B rows)
  __shared__ unsigned short Bs[64][34];  // [n][k] transposed, bf16

  const int t = threadIdx.x;
  const int lane = t & 31, wave = t >> 5;
  const int lh = lane >> 4, ln = lane & 15;
  const int m0 = blockIdx.y * 64, n0 = blockIdx.x * 64;
  const int wm = (wave >> 1) * 32, wn = (wave & 1) * 32;
  const unsigned int* Wu = (const unsigned int*)W;

  v8f acc[2][2] = {};

  for (int k0 = 0; k0 < 512; k0 += 32) {
    __syncthreads();
    // stage A: 64x32 fp32 -> bf16 (float4 loads, packed dword LDS stores)
#pragma unroll
    for (int j = 0; j < 4; ++j) {
      int e = t + 128 * j;              // 512 float4 groups
      int row = e >> 3, c4 = e & 7;
      const float4 f = *(const float4*)(X + (m0 + row) * 512 + k0 + c4 * 4);
      unsigned int* p = (unsigned int*)&As[row][c4 * 4];
      p[0] = (unsigned int)f32_to_bf16(f.x) | ((unsigned int)f32_to_bf16(f.y) << 16);
      p[1] = (unsigned int)f32_to_bf16(f.z) | ((unsigned int)f32_to_bf16(f.w) << 16);
    }
    // stage B transposed: 32x64 bf16 -> Bs[n][k]
#pragma unroll
    for (int j = 0; j < 8; ++j) {
      int e = t + 128 * j;              // 1024 dwords
      int kr = e >> 5, cu = e & 31;
      unsigned int u = Wu[(k0 + kr) * 384 + (n0 >> 1) + cu];
      Bs[cu * 2][kr]     = (unsigned short)(u & 0xFFFFu);
      Bs[cu * 2 + 1][kr] = (unsigned short)(u >> 16);
    }
    if (k0 + 32 < 512) {                // gfx1250 global_prefetch_b8
      __builtin_prefetch(X + m0 * 512 + k0 + 32, 0, 0);
      __builtin_prefetch(W + (size_t)(k0 + 32) * 768 + n0, 0, 0);
    }
    __syncthreads();

    Frag16 a[2], b[2];
#pragma unroll
    for (int i = 0; i < 2; ++i) {
      int row = wm + i * 16 + ln;       // A: M = lane&15, K-half by lane>>4
#pragma unroll
      for (int v = 0; v < 8; ++v) {
        int kk = (v < 4 ? 2 * v : 16 + 2 * (v - 4)) + lh * 8;
        a[i].u[v] = *(const unsigned int*)&As[row][kk];
      }
      int col = wn + i * 16 + ln;       // B: N = lane&15, K-half by lane>>4
#pragma unroll
      for (int v = 0; v < 8; ++v)
        b[i].u[v] = *(const unsigned int*)&Bs[col][(v < 4 ? 2 * v : 16 + 2 * (v - 4)) + lh * 8];
    }
#pragma unroll
    for (int i = 0; i < 2; ++i)
#pragma unroll
      for (int jn = 0; jn < 2; ++jn)
        acc[i][jn] = __builtin_amdgcn_wmma_f32_16x16x32_bf16(
            false, a[i].v, false, b[jn].v, (short)0, acc[i][jn], false, false);
  }

#pragma unroll
  for (int i = 0; i < 2; ++i)
#pragma unroll
    for (int jn = 0; jn < 2; ++jn)
#pragma unroll
      for (int r = 0; r < 8; ++r) {     // D: M = 8*half + r, N = lane&15
        int row = m0 + wm + i * 16 + lh * 8 + r;
        int col = n0 + wn + jn * 16 + ln;
        Y[row * 768 + col] = f32_to_bf16(acc[i][jn][r]);
      }
}

// ---------------------------------------------------------------------------
// Kernel 3: block-diagonal attention. One WG per (b, h, block, 64-query slice).
// Q[64x32], K[256x32], V^T[32x256] staged in LDS (bf16). d=32 => Q.K^T is one
// WMMA (K=32) per 16x16 score tile. Softmax rows in LDS, then P@V (8 K-steps).
// ---------------------------------------------------------------------------
__global__ __launch_bounds__(128) void attn_kernel(
    const unsigned short* __restrict__ QKV,   // [16384][768] bf16
    unsigned short* __restrict__ O) {         // [16384][256] bf16
  extern __shared__ unsigned char smem[];
  unsigned short* Qs = (unsigned short*)smem;          // [64][34]
  unsigned short* Ks = Qs + 64 * 34;                   // [256][34]
  unsigned short* Vt = Ks + 256 * 34;                  // [32][258] (d-major)
  float* Sc = (float*)(smem + (64 * 34 + 256 * 34 + 32 * 258) * 2); // [64][256]
  unsigned short* P = (unsigned short*)(Sc + 64 * 256);             // [64][258]

  const int t = threadIdx.x, lane = t & 31, wave = t >> 5;
  const int lh = lane >> 4, ln = lane & 15;
  const int idx = blockIdx.x;
  const int qt = idx & 3, blk = (idx >> 2) & 15;
  const int h = (idx >> 6) & 7, b = idx >> 9;
  const int tb = b * 4096 + blk * 256;      // block token base
  const int qb0 = tb + qt * 64;             // first query token of this WG
  const int hc = h * 32;                    // head column offset

  const unsigned int* Qu = (const unsigned int*)QKV;  // dword view (2 bf16)

  // stage Q (64 rows x 16 dwords)
#pragma unroll
  for (int j = 0; j < 8; ++j) {
    int e = t + 128 * j, row = e >> 4, cu = e & 15;
    *(unsigned int*)&Qs[row * 34 + cu * 2] =
        Qu[(qb0 + row) * 384 + (hc >> 1) + cu];
  }
  // stage K (256 rows x 16 dwords), natural [key][d]
#pragma unroll 4
  for (int j = 0; j < 32; ++j) {
    int e = t + 128 * j, row = e >> 4, cu = e & 15;
    *(unsigned int*)&Ks[row * 34 + cu * 2] =
        Qu[(tb + row) * 384 + ((256 + hc) >> 1) + cu];
  }
  // stage V transposed -> Vt[d][key]
#pragma unroll 4
  for (int j = 0; j < 32; ++j) {
    int e = t + 128 * j, row = e >> 4, cu = e & 15;
    unsigned int u = Qu[(tb + row) * 384 + ((512 + hc) >> 1) + cu];
    Vt[(cu * 2) * 258 + row]     = (unsigned short)(u & 0xFFFFu);
    Vt[(cu * 2 + 1) * 258 + row] = (unsigned short)(u >> 16);
  }
  __syncthreads();

  // ---- scores: Q K^T * scale  (wave owns 16 query rows) ----
  Frag16 qa;
  {
    int row = wave * 16 + ln;
#pragma unroll
    for (int v = 0; v < 8; ++v) {
      int kk = (v < 4 ? 2 * v : 16 + 2 * (v - 4)) + lh * 8;
      qa.u[v] = *(const unsigned int*)&Qs[row * 34 + kk];
    }
  }
  for (int nt = 0; nt < 16; ++nt) {
    Frag16 kb;
    int col = nt * 16 + ln;
#pragma unroll
    for (int v = 0; v < 8; ++v)
      kb.u[v] = *(const unsigned int*)&Ks[col * 34 + (v < 4 ? 2 * v : 16 + 2 * (v - 4)) + lh * 8];
    v8f s = {};
    s = __builtin_amdgcn_wmma_f32_16x16x32_bf16(false, qa.v, false, kb.v,
                                                (short)0, s, false, false);
#pragma unroll
    for (int r = 0; r < 8; ++r)
      Sc[(wave * 16 + lh * 8 + r) * 256 + nt * 16 + ln] = s[r] * 0.125f;
  }
  __syncthreads();

  // ---- softmax: one lane per query row ----
  if (lane < 16) {
    int row = wave * 16 + lane;
    float* sr = Sc + row * 256;
    float m = -3.402823e38f;
    for (int k = 0; k < 256; ++k) m = fmaxf(m, sr[k]);
    float sum = 0.f;
    for (int k = 0; k < 256; ++k) {
      float e = __expf(sr[k] - m);
      sr[k] = e;
      sum += e;
    }
    float inv = 1.f / sum;
    unsigned short* pr = P + row * 258;
    for (int k = 0; k < 256; ++k) pr[k] = f32_to_bf16(sr[k] * inv);
  }
  __syncthreads();

  // ---- O = P @ V ----
#pragma unroll
  for (int nt = 0; nt < 2; ++nt) {
    v8f acc = {};
    for (int kt = 0; kt < 8; ++kt) {
      Frag16 pa, vb;
      int row = wave * 16 + ln;
#pragma unroll
      for (int v = 0; v < 8; ++v) {
        int kk = kt * 32 + (v < 4 ? 2 * v : 16 + 2 * (v - 4)) + lh * 8;
        pa.u[v] = *(const unsigned int*)&P[row * 258 + kk];
      }
      int col = nt * 16 + ln;
#pragma unroll
      for (int v = 0; v < 8; ++v)
        vb.u[v] = *(const unsigned int*)&Vt[col * 258 + kt * 32 + (v < 4 ? 2 * v : 16 + 2 * (v - 4)) + lh * 8];
      acc = __builtin_amdgcn_wmma_f32_16x16x32_bf16(false, pa.v, false, vb.v,
                                                    (short)0, acc, false, false);
    }
#pragma unroll
    for (int r = 0; r < 8; ++r) {
      int token = qb0 + wave * 16 + lh * 8 + r;
      int col = hc + nt * 16 + ln;
      O[token * 256 + col] = f32_to_bf16(acc[r]);
    }
  }
}

// ---------------------------------------------------------------------------
// Kernel 4: out[16384,512] (f32) = o[16384,256] (bf16) @ PW[256,512] (bf16) + b
// ---------------------------------------------------------------------------
__global__ __launch_bounds__(128) void gemm_proj(
    const unsigned short* __restrict__ A, const unsigned short* __restrict__ W,
    const float* __restrict__ bias, float* __restrict__ Y) {
  __shared__ unsigned short As[64][36];
  __shared__ unsigned short Bs[64][34];

  const int t = threadIdx.x;
  const int lane = t & 31, wave = t >> 5;
  const int lh = lane >> 4, ln = lane & 15;
  const int m0 = blockIdx.y * 64, n0 = blockIdx.x * 64;
  const int wm = (wave >> 1) * 32, wn = (wave & 1) * 32;
  const unsigned int* Au = (const unsigned int*)A;
  const unsigned int* Wu = (const unsigned int*)W;

  v8f acc[2][2] = {};

  for (int k0 = 0; k0 < 256; k0 += 32) {
    __syncthreads();
#pragma unroll
    for (int j = 0; j < 8; ++j) {       // A: 64x32 bf16 = 1024 dwords
      int e = t + 128 * j, row = e >> 4, cu = e & 15;
      *(unsigned int*)&As[row][cu * 2] = Au[(m0 + row) * 128 + (k0 >> 1) + cu];
    }
#pragma unroll
    for (int j = 0; j < 8; ++j) {       // B: 32x64 -> Bs[n][k]
      int e = t + 128 * j, kr = e >> 5, cu = e & 31;
      unsigned int u = Wu[(k0 + kr) * 256 + (n0 >> 1) + cu];
      Bs[cu * 2][kr]     = (unsigned short)(u & 0xFFFFu);
      Bs[cu * 2 + 1][kr] = (unsigned short)(u >> 16);
    }
    if (k0 + 32 < 256) {
      __builtin_prefetch(A + (size_t)m0 * 256 + k0 + 32, 0, 0);
      __builtin_prefetch(W + (size_t)(k0 + 32) * 512 + n0, 0, 0);
    }
    __syncthreads();

    Frag16 a[2], b[2];
#pragma unroll
    for (int i = 0; i < 2; ++i) {
      int row = wm + i * 16 + ln;
#pragma unroll
      for (int v = 0; v < 8; ++v) {
        int kk = (v < 4 ? 2 * v : 16 + 2 * (v - 4)) + lh * 8;
        a[i].u[v] = *(const unsigned int*)&As[row][kk];
      }
      int col = wn + i * 16 + ln;
#pragma unroll
      for (int v = 0; v < 8; ++v)
        b[i].u[v] = *(const unsigned int*)&Bs[col][(v < 4 ? 2 * v : 16 + 2 * (v - 4)) + lh * 8];
    }
#pragma unroll
    for (int i = 0; i < 2; ++i)
#pragma unroll
      for (int jn = 0; jn < 2; ++jn)
        acc[i][jn] = __builtin_amdgcn_wmma_f32_16x16x32_bf16(
            false, a[i].v, false, b[jn].v, (short)0, acc[i][jn], false, false);
  }

#pragma unroll
  for (int i = 0; i < 2; ++i)
#pragma unroll
    for (int jn = 0; jn < 2; ++jn)
#pragma unroll
      for (int r = 0; r < 8; ++r) {
        int row = m0 + wm + i * 16 + lh * 8 + r;
        int col = n0 + wn + jn * 16 + ln;
        Y[row * 512 + col] = acc[i][jn][r] + bias[col];
      }
}

// ---------------------------------------------------------------------------
// Host launcher
// ---------------------------------------------------------------------------
extern "C" void kernel_launch(void* const* d_in, const int* in_sizes, int n_in,
                              void* d_out, int out_size, void* d_ws,
                              size_t ws_size, hipStream_t stream) {
  (void)in_sizes; (void)n_in; (void)out_size; (void)ws_size;
  const float* x        = (const float*)d_in[0];   // [4,4096,512]
  const float* reduce_w = (const float*)d_in[1];   // [512,256]
  const float* qkv_w    = (const float*)d_in[2];   // [256,768]
  const float* proj_w   = (const float*)d_in[3];   // [256,512]
  const float* proj_b   = (const float*)d_in[4];   // [512]
  float* out = (float*)d_out;                      // [4,4096,512]

  char* ws = (char*)d_ws;
  unsigned short* Wf   = (unsigned short*)(ws);              // 512*768  bf16
  unsigned short* PW   = (unsigned short*)(ws + 786432);     // 256*512  bf16
  unsigned short* QKVb = (unsigned short*)(ws + 1048576);    // 16384*768 bf16
  unsigned short* Ob   = (unsigned short*)(ws + 26214400);   // 16384*256 bf16

  prep_weights<<<2048, 256, 0, stream>>>(reduce_w, qkv_w, proj_w, Wf, PW);
  gemm_qkv<<<dim3(12, 256), 128, 0, stream>>>(x, Wf, QKVb);
  size_t smem = (size_t)(64 * 34 + 256 * 34 + 32 * 258) * 2 +
                (size_t)64 * 256 * 4 + (size_t)64 * 258 * 2;   // 136832 B
  attn_kernel<<<2048, 128, smem, stream>>>(QKVb, Ob);
  gemm_proj<<<dim3(8, 256), 128, 0, stream>>>(Ob, PW, proj_b, out);
}
